// TransformerDiffuser_77524159692895
// MI455X (gfx1250) — matrix operand order repaired and checked by reading, not compiled, not run on hardware
//
#include <hip/hip_runtime.h>
#include <hip/hip_bf16.h>
#include <math.h>

// ---------------------------------------------------------------------------
// Model dims (fixed by the reference)
// ---------------------------------------------------------------------------
// B=8, C=2, FDIM=128, T=1024, PF=16, PT=8, D=768, H=12, DH=64, L=8
// FP=8, TP=128, N=1024 tokens, PV=256, M = B*N = 8192 rows
// ---------------------------------------------------------------------------

typedef __attribute__((ext_vector_type(16))) __bf16 v16bf;
typedef __attribute__((ext_vector_type(8)))  float  v8f;

#define RES_SCALE 0.70710678118654752f
#define LN10K     9.2103403719761836f   // ln(10000)

__device__ __forceinline__ unsigned short f2bf(float f) {
  unsigned int u = __float_as_uint(f);
  u += 0x7FFFu + ((u >> 16) & 1u);           // round-to-nearest-even
  return (unsigned short)(u >> 16);
}

union Frag { uint4 q[2]; v16bf v; };

// 16-bit A fragment (16x32): lanes 0-15 -> K 0-7 & 16-23 ; lanes 16-31 -> K 8-15 & 24-31
__device__ __forceinline__ v16bf load_fragA(const unsigned short* row, int kt, int lane) {
  const int off = kt + ((lane & 16) ? 8 : 0);
  Frag f;
  f.q[0] = *(const uint4*)(row + off);
  f.q[1] = *(const uint4*)(row + off + 16);
  return f.v;
}
// 16-bit B fragment (32x16): lanes 0-15 -> K 0-7 & 16-23 ; lanes 16-31 -> K 8-15 & 24-31
__device__ __forceinline__ v16bf load_fragB(const unsigned short* row, int kt, int lane) {
  const int off = kt + ((lane & 16) ? 8 : 0);
  Frag f;
  f.q[0] = *(const uint4*)(row + off);
  f.q[1] = *(const uint4*)(row + off + 16);
  return f.v;
}

// ---------------------------------------------------------------------------
// Generic tiled WMMA GEMM: Out(MxNc) = epi(A(MxK) * Bw(KxNc) + bias)
// BM=BN=128, BK=32, 256 threads = 8 waves, wave -> 64x32 (4x2 WMMA tiles)
// A and Bw are f32 in global memory, converted to bf16 while staging to LDS.
// ---------------------------------------------------------------------------
enum { EPI_PLAIN = 0, EPI_ROWTAB = 1, EPI_GELU = 2, EPI_GATERES = 3 };

template<int EPI>
__global__ __launch_bounds__(256)
void gemm_bf16(const float* __restrict__ A, const float* __restrict__ Bw,
               const float* __restrict__ bias, float* __restrict__ Out,
               int M, int Nc, int K, int ldo,
               const float* __restrict__ rowtab,            // EPI_ROWTAB: (1024 x Nc)
               const float* __restrict__ gate, int gstride,  // EPI_GATERES: gate[b][col]
               float gscale)
{
  __shared__ alignas(16) unsigned short lsA[128][40];  // 32 + 8 pad
  __shared__ alignas(16) unsigned short lsB[128][40];  // transposed: [n][k]

  const int t    = threadIdx.x;
  const int lane = t & 31;
  const int wid  = t >> 5;
  const int m0   = blockIdx.y * 128;
  const int n0   = blockIdx.x * 128;
  const int mw   = (wid >> 2) * 64;
  const int nw   = (wid & 3) * 32;
  const int lr   = lane & 15;

  const v8f vzero = {0.f,0.f,0.f,0.f,0.f,0.f,0.f,0.f};
  v8f acc[4][2];
#pragma unroll
  for (int mi = 0; mi < 4; ++mi)
#pragma unroll
    for (int ni = 0; ni < 2; ++ni) acc[mi][ni] = vzero;

  for (int kt = 0; kt < K; kt += 32) {
    // stage A tile 128x32 (f32 -> bf16)
    {
      const int r = t >> 3;
      const int c = (t & 7) << 2;
#pragma unroll
      for (int i = 0; i < 4; ++i) {
        const float4 v = *(const float4*)(A + (size_t)(m0 + r + i * 32) * K + (kt + c));
        unsigned short* p = &lsA[r + i * 32][c];
        p[0] = f2bf(v.x); p[1] = f2bf(v.y); p[2] = f2bf(v.z); p[3] = f2bf(v.w);
      }
    }
    // stage B tile 32x128 transposed -> lsB[n][k]
    {
      const int kr = t >> 5;
      const int c  = (t & 31) << 2;
#pragma unroll
      for (int i = 0; i < 4; ++i) {
        const float4 v = *(const float4*)(Bw + (size_t)(kt + kr + i * 8) * Nc + (n0 + c));
        lsB[c + 0][kr + i * 8] = f2bf(v.x);
        lsB[c + 1][kr + i * 8] = f2bf(v.y);
        lsB[c + 2][kr + i * 8] = f2bf(v.z);
        lsB[c + 3][kr + i * 8] = f2bf(v.w);
      }
    }
    if (kt + 32 < K) {  // emits global_prefetch_b8 for next K tile
      __builtin_prefetch(A + (size_t)(m0 + (t >> 3)) * K + kt + 32, 0, 1);
      __builtin_prefetch(Bw + (size_t)(kt + 32 + (t >> 5)) * Nc + n0 + ((t & 31) << 2), 0, 1);
    }
    __syncthreads();

    v16bf af[4];
#pragma unroll
    for (int mi = 0; mi < 4; ++mi)
      af[mi] = load_fragA(&lsA[mw + mi * 16 + lr][0], 0, lane);
#pragma unroll
    for (int ni = 0; ni < 2; ++ni) {
      const v16bf bf = load_fragB(&lsB[nw + ni * 16 + lr][0], 0, lane);
#pragma unroll
      for (int mi = 0; mi < 4; ++mi)
        acc[mi][ni] = __builtin_amdgcn_wmma_f32_16x16x32_bf16(
            false, af[mi], false, bf, (short)0, acc[mi][ni], false, false);
    }
    __syncthreads();
  }

  // epilogue (C layout: vgpr j = row j (lanes 0-15) / row j+8 (lanes 16-31))
  const int rbase = m0 + mw + ((lane & 16) ? 8 : 0);
#pragma unroll
  for (int ni = 0; ni < 2; ++ni) {
    const int col = n0 + nw + ni * 16 + lr;
    const float bv = bias[col];
#pragma unroll
    for (int mi = 0; mi < 4; ++mi) {
#pragma unroll
      for (int j = 0; j < 8; ++j) {
        const int row = rbase + mi * 16 + j;
        float v = acc[mi][ni][j] + bv;
        const size_t oidx = (size_t)row * ldo + col;
        if (EPI == EPI_ROWTAB) {
          Out[oidx] = v + rowtab[(size_t)(row & 1023) * Nc + col];
        } else if (EPI == EPI_GELU) {
          Out[oidx] = 0.5f * v * (1.0f + erff(v * 0.70710678f));
        } else if (EPI == EPI_GATERES) {
          Out[oidx] += gscale * gate[(size_t)(row >> 10) * gstride + col] * v;
        } else {
          Out[oidx] = v;
        }
      }
    }
  }
}

// ---------------------------------------------------------------------------
// Flash attention: per (b,h) & 128-query block; KV blocks of 128; DH=64.
// S = Q K^T via WMMA, online softmax (f32, 16-lane shfl reductions),
// P routed through per-wave LDS tile -> WMMA P*V.  Output Y in (B,N,D).
// ---------------------------------------------------------------------------
__global__ __launch_bounds__(256)
void flash_attn(const float* __restrict__ Q, const float* __restrict__ Kx,
                const float* __restrict__ Vx, float* __restrict__ Y, float scale)
{
  __shared__ alignas(16) unsigned short sQ[128][72];      // [qrow][dh]
  __shared__ alignas(16) unsigned short sK[128][72];      // [kvrow][dh]  (== B^T for QK^T)
  __shared__ alignas(16) unsigned short sV[64][136];      // [dh][kvrow]  (V transposed)
  __shared__ alignas(16) unsigned short sP[8][16][136];   // per-wave P tile 16x128

  const int t = threadIdx.x, lane = t & 31, wid = t >> 5, lr = lane & 15;
  const int bh = blockIdx.y;          // 0..95
  const int qb = blockIdx.x;          // 0..7
  const int b  = bh / 12, hh = bh % 12;
  const size_t base = (size_t)bh * 1024 * 64;

  // stage Q block once
  {
    const int r = t >> 1, cb = (t & 1) * 32;
#pragma unroll
    for (int i = 0; i < 8; ++i) {
      const float4 v = *(const float4*)(Q + base + (size_t)(qb * 128 + r) * 64 + cb + i * 4);
      unsigned short* p = &sQ[r][cb + i * 4];
      p[0] = f2bf(v.x); p[1] = f2bf(v.y); p[2] = f2bf(v.z); p[3] = f2bf(v.w);
    }
  }

  const v8f vzero = {0.f,0.f,0.f,0.f,0.f,0.f,0.f,0.f};
  v8f o[4];
#pragma unroll
  for (int i = 0; i < 4; ++i) o[i] = vzero;
  float mprev[8], lsum[8];
#pragma unroll
  for (int j = 0; j < 8; ++j) { mprev[j] = -3.0e38f; lsum[j] = 0.f; }

  for (int kb = 0; kb < 8; ++kb) {
    __syncthreads();
    // stage K rows and V transposed
    {
      const int r = t >> 1, cb = (t & 1) * 32;
#pragma unroll
      for (int i = 0; i < 8; ++i) {
        const float4 v = *(const float4*)(Kx + base + (size_t)(kb * 128 + r) * 64 + cb + i * 4);
        unsigned short* p = &sK[r][cb + i * 4];
        p[0] = f2bf(v.x); p[1] = f2bf(v.y); p[2] = f2bf(v.z); p[3] = f2bf(v.w);
        const float4 w = *(const float4*)(Vx + base + (size_t)(kb * 128 + r) * 64 + cb + i * 4);
        sV[cb + i * 4 + 0][r] = f2bf(w.x);
        sV[cb + i * 4 + 1][r] = f2bf(w.y);
        sV[cb + i * 4 + 2][r] = f2bf(w.z);
        sV[cb + i * 4 + 3][r] = f2bf(w.w);
      }
    }
    __syncthreads();

    // S tile: wave rows [wid*16, wid*16+16), cols 0..127
    v8f s[8];
#pragma unroll
    for (int ni = 0; ni < 8; ++ni) s[ni] = vzero;
#pragma unroll
    for (int kt = 0; kt < 64; kt += 32) {
      const v16bf a = load_fragA(&sQ[wid * 16 + lr][0], kt, lane);
#pragma unroll
      for (int ni = 0; ni < 8; ++ni) {
        const v16bf bf = load_fragB(&sK[ni * 16 + lr][0], kt, lane);
        s[ni] = __builtin_amdgcn_wmma_f32_16x16x32_bf16(
            false, a, false, bf, (short)0, s[ni], false, false);
      }
    }

    // online softmax per row (row spread over 16 lanes of a half-wave)
#pragma unroll
    for (int j = 0; j < 8; ++j) {
      float rm = -3.0e38f;
#pragma unroll
      for (int ni = 0; ni < 8; ++ni) { s[ni][j] *= scale; rm = fmaxf(rm, s[ni][j]); }
#pragma unroll
      for (int off = 8; off >= 1; off >>= 1) rm = fmaxf(rm, __shfl_xor(rm, off, 16));
      const float mnew  = fmaxf(mprev[j], rm);
      const float alpha = __expf(mprev[j] - mnew);
      float rs = 0.f;
#pragma unroll
      for (int ni = 0; ni < 8; ++ni) {
        const float p = __expf(s[ni][j] - mnew);
        s[ni][j] = p; rs += p;
      }
#pragma unroll
      for (int off = 8; off >= 1; off >>= 1) rs += __shfl_xor(rs, off, 16);
      lsum[j]  = lsum[j] * alpha + rs;
      mprev[j] = mnew;
#pragma unroll
      for (int oi = 0; oi < 4; ++oi) o[oi][j] *= alpha;
    }

    // write P (C layout) to per-wave LDS tile as bf16
    const int rj = (lane & 16) ? 8 : 0;
#pragma unroll
    for (int ni = 0; ni < 8; ++ni)
#pragma unroll
      for (int j = 0; j < 8; ++j)
        sP[wid][j + rj][ni * 16 + lr] = f2bf(s[ni][j]);
    asm volatile("s_wait_dscnt 0" ::: "memory");   // LDS RAW fence (same wave)

    // O += P * V
#pragma unroll
    for (int kt = 0; kt < 128; kt += 32) {
      const v16bf a = load_fragA(&sP[wid][lr][0], kt, lane);
#pragma unroll
      for (int ni = 0; ni < 4; ++ni) {
        const v16bf bf = load_fragB(&sV[ni * 16 + lr][0], kt, lane);
        o[ni] = __builtin_amdgcn_wmma_f32_16x16x32_bf16(
            false, a, false, bf, (short)0, o[ni], false, false);
      }
    }
  }

  // normalize and write to Y (B,N,D) with head offset
  const int rj = (lane & 16) ? 8 : 0;
#pragma unroll
  for (int ni = 0; ni < 4; ++ni) {
    const int col = hh * 64 + ni * 16 + lr;
#pragma unroll
    for (int j = 0; j < 8; ++j) {
      const int n = qb * 128 + wid * 16 + j + rj;
      Y[(size_t)(b * 1024 + n) * 768 + col] = o[ni][j] / lsum[j];
    }
  }
}

// ---------------------------------------------------------------------------
// Small / elementwise kernels
// ---------------------------------------------------------------------------
__global__ void ddpm_emb_kernel(const int* __restrict__ tt, float* __restrict__ e0) {
  const int id = blockIdx.x * 256 + threadIdx.x;   // 8*768
  if (id >= 8 * 768) return;
  const int bb = id / 768, d = id % 768;
  const int i = (d < 384) ? d : d - 384;
  const float a = (float)tt[bb] * __expf(-LN10K * (float)i / 383.f);
  e0[id] = (d < 384) ? __sinf(a) : __cosf(a);
}

template<bool SILU_A, bool SILU_OUT>
__global__ void small_mm(const float* __restrict__ A, const float* __restrict__ Arow,
                         const float* __restrict__ W, const float* __restrict__ bias,
                         float* __restrict__ Out, int R, int Cc, int Kd)
{
  const int id = blockIdx.x * 256 + threadIdx.x;
  if (id >= R * Cc) return;
  const int r = id / Cc, c = id % Cc;
  const float* ar = A + (size_t)r * Kd;
  float sum = 0.f;
  for (int k = 0; k < Kd; ++k) {
    float av = ar[k];
    if (Arow) av += Arow[k];
    if (SILU_A) av = av / (1.f + __expf(-av));
    sum += av * W[(size_t)k * Cc + c];
  }
  float v = sum + bias[c];
  if (SILU_OUT) v = v / (1.f + __expf(-v));
  Out[id] = v;
}

__global__ void posenc_kernel(float* __restrict__ pe) {
  const int id = blockIdx.x * 256 + threadIdx.x;   // 1024*768
  if (id >= 1024 * 768) return;
  const int n = id / 768, d = id % 768;
  const int fp = n >> 7, tp = n & 127;
  const int j = (d < 384) ? d : d - 384;
  const float pos = (d < 384) ? (float)fp : (float)tp;
  const int i = (j < 192) ? j : j - 192;
  const float ang = pos * __expf(-LN10K * (float)i / 191.f);
  pe[id] = (j < 192) ? __sinf(ang) : __cosf(ang);
}

__global__ void patch_gather_kernel(const float* __restrict__ x, float* __restrict__ Ap) {
  const int id = blockIdx.x * 256 + threadIdx.x;   // 8192*256
  const int pv = id & 255, row = id >> 8;
  const int b = row >> 10, n = row & 1023;
  const int fp = n >> 7, tp = n & 127;
  const int c = pv >> 7, rem = pv & 127, pf = rem >> 3, pt = rem & 7;
  Ap[id] = x[(((size_t)(b * 2 + c) * 128 + fp * 16 + pf) << 10) + tp * 8 + pt];
}

__global__ void cat_ttok_kernel(const float* __restrict__ ttok, float* __restrict__ cat) {
  const int id = blockIdx.x * 256 + threadIdx.x;   // 8192*768
  const int r = id / 768, c = id % 768;
  cat[(size_t)r * 1536 + 768 + c] = ttok[(r >> 10) * 768 + c];
}

__global__ __launch_bounds__(256)
void ln_mod_kernel(const float* __restrict__ X, const float* __restrict__ w,
                   const float* __restrict__ bb, const float* __restrict__ cond,
                   float* __restrict__ Outp)
{
  __shared__ float red[256];
  const int r = blockIdx.x, t = threadIdx.x;
  const float* xr = X + (size_t)r * 768;
  const float x0 = xr[t], x1 = xr[t + 256], x2 = xr[t + 512];
  red[t] = x0 + x1 + x2;
  __syncthreads();
  for (int st = 128; st > 0; st >>= 1) { if (t < st) red[t] += red[t + st]; __syncthreads(); }
  const float mu = red[0] * (1.f / 768.f);
  __syncthreads();
  const float d0 = x0 - mu, d1 = x1 - mu, d2 = x2 - mu;
  red[t] = d0 * d0 + d1 * d1 + d2 * d2;
  __syncthreads();
  for (int st = 128; st > 0; st >>= 1) { if (t < st) red[t] += red[t + st]; __syncthreads(); }
  const float rstd = rsqrtf(red[0] * (1.f / 768.f) + 1e-5f);
  const int bi = r >> 10;
  const float* sh = cond + (size_t)bi * 2304;
  const float* sc = sh + 768;
  float* orow = Outp + (size_t)r * 768;
  orow[t      ] = (d0 * rstd * w[t      ] + bb[t      ]) * (1.f + sc[t      ]) + sh[t      ];
  orow[t + 256] = (d1 * rstd * w[t + 256] + bb[t + 256]) * (1.f + sc[t + 256]) + sh[t + 256];
  orow[t + 512] = (d2 * rstd * w[t + 512] + bb[t + 512]) * (1.f + sc[t + 512]) + sh[t + 512];
}

__global__ void qkv_rope_kernel(const float* __restrict__ qkv, float* __restrict__ q,
                                float* __restrict__ k, float* __restrict__ v)
{
  const int id = blockIdx.x * 256 + threadIdx.x;   // 96*1024*64
  const int d  = id & 63;
  const int n  = (id >> 6) & 1023;
  const int bh = id >> 16;
  const int b = bh / 12, hh = bh % 12;
  const size_t rowb = (size_t)(b * 1024 + n) * 2304;
  const int cq = hh * 64 + d;
  const float qx = qkv[rowb + cq];
  const float kx = qkv[rowb + 768 + cq];
  const float vx = qkv[rowb + 1536 + cq];
  const int fi = d & 31;
  const float ang = (float)n * __expf(-LN10K * (float)fi / 32.f);
  const float cs = __cosf(ang), sn = __sinf(ang);
  const int dp = (d & 1) ? (d - 1) : (d + 1);
  const float qp = qkv[rowb + hh * 64 + dp];
  const float kp = qkv[rowb + 768 + hh * 64 + dp];
  const float yq = (d & 1) ? qp : -qp;
  const float yk = (d & 1) ? kp : -kp;
  q[id] = qx * cs + yq * sn;
  k[id] = kx * cs + yk * sn;
  v[id] = vx;
}

__global__ void final_out_kernel(const float* __restrict__ x, const float* __restrict__ pat,
                                 const float* __restrict__ gain, float* __restrict__ out)
{
  const int id = blockIdx.x * 256 + threadIdx.x;   // 8*2*128*1024
  const int tt = id & 1023;
  const int f  = (id >> 10) & 127;
  const int c  = (id >> 17) & 1;
  const int b  = id >> 18;
  const int fp = f >> 4, pf = f & 15, tp = tt >> 3, pt = tt & 7;
  const int n  = fp * 128 + tp;
  const int pv = c * 128 + pf * 8 + pt;
  out[id] = x[id] + gain[c] * pat[(size_t)(b * 1024 + n) * 256 + pv];
}

// ---------------------------------------------------------------------------
// Driver
// ---------------------------------------------------------------------------
extern "C" void kernel_launch(void* const* d_in, const int* in_sizes, int n_in,
                              void* d_out, int out_size, void* d_ws, size_t ws_size,
                              hipStream_t stream)
{
  (void)in_sizes; (void)n_in; (void)out_size; (void)ws_size;
  const float* x            = (const float*)d_in[0];
  const int*   tarr         = (const int*)  d_in[1];
  const float* patch_w      = (const float*)d_in[2];
  const float* patch_b      = (const float*)d_in[3];
  const float* tmlp_w1      = (const float*)d_in[4];
  const float* tmlp_b1      = (const float*)d_in[5];
  const float* tmlp_w2      = (const float*)d_in[6];
  const float* tmlp_b2      = (const float*)d_in[7];
  const float* ttok_w       = (const float*)d_in[8];
  const float* ttok_b       = (const float*)d_in[9];
  const float* tokin_w      = (const float*)d_in[10];
  const float* tokin_b      = (const float*)d_in[11];
  const float* block_embed  = (const float*)d_in[12];
  const float* ln1_w        = (const float*)d_in[13];
  const float* ln1_b        = (const float*)d_in[14];
  const float* ln2_w        = (const float*)d_in[15];
  const float* ln2_b        = (const float*)d_in[16];
  const float* qkv_w        = (const float*)d_in[17];
  const float* qkv_b        = (const float*)d_in[18];
  const float* attn_proj_w  = (const float*)d_in[19];
  const float* attn_proj_b  = (const float*)d_in[20];
  const float* mlp_w1       = (const float*)d_in[21];
  const float* mlp_b1       = (const float*)d_in[22];
  const float* mlp_w2       = (const float*)d_in[23];
  const float* mlp_b2       = (const float*)d_in[24];
  const float* cond1_w      = (const float*)d_in[25];
  const float* cond1_b      = (const float*)d_in[26];
  const float* cond2_w      = (const float*)d_in[27];
  const float* cond2_b      = (const float*)d_in[28];
  const float* unembed_w    = (const float*)d_in[29];
  const float* unembed_b    = (const float*)d_in[30];
  const float* out_gain     = (const float*)d_in[31];
  float* out = (float*)d_out;

  // workspace layout (floats), with reuse
  float* w = (float*)d_ws;
  size_t off = 0;
  auto alloc = [&](size_t n) { float* p = w + off; off += n; return p; };
  float* f_h    = alloc(8192ull * 768);
  float* f_hn   = alloc(8192ull * 768);
  float* f_cat  = alloc(8192ull * 1536);   // later reused as q (lo half) / k (hi half)
  float* f_big  = alloc(8192ull * 3072);   // qkv (18.9M) and mlp hidden (25.2M) union
  float* f_v    = alloc(8192ull * 768);
  float* f_y    = alloc(8192ull * 768);
  float* f_pat  = alloc(8192ull * 256);    // patch A-matrix, then unembed output
  float* f_pe   = alloc(1024ull * 768);
  float* f_e0   = alloc(8 * 768);
  float* f_e1   = alloc(8 * 768);
  float* f_temb = alloc(8 * 768);
  float* f_ttok = alloc(8 * 768);
  float* f_c1   = alloc(8 * 2304);
  float* f_c2   = alloc(8 * 2304);
  float* f_q    = f_cat;
  float* f_k    = f_cat + 8192ull * 768;
  float* f_qkv  = f_big;
  float* f_m1   = f_big;
  float* f_Ap   = f_pat;

  const dim3 blk(256);

  // --- time-embedding chain (tiny) ---
  ddpm_emb_kernel<<<24, blk, 0, stream>>>(tarr, f_e0);
  small_mm<false, true ><<<24, blk, 0, stream>>>(f_e0,   nullptr, tmlp_w1, tmlp_b1, f_e1,   8, 768, 768);
  small_mm<false, false><<<24, blk, 0, stream>>>(f_e1,   nullptr, tmlp_w2, tmlp_b2, f_temb, 8, 768, 768);
  small_mm<true,  false><<<24, blk, 0, stream>>>(f_temb, nullptr, ttok_w,  ttok_b,  f_ttok, 8, 768, 768);

  // --- patchify + token input ---
  posenc_kernel<<<3072, blk, 0, stream>>>(f_pe);
  patch_gather_kernel<<<8192, blk, 0, stream>>>(x, f_Ap);
  cat_ttok_kernel<<<24576, blk, 0, stream>>>(f_ttok, f_cat);
  gemm_bf16<EPI_ROWTAB><<<dim3(6, 64), blk, 0, stream>>>(
      f_Ap, patch_w, patch_b, f_cat, 8192, 768, 256, 1536, f_pe, nullptr, 0, 0.f);
  gemm_bf16<EPI_PLAIN><<<dim3(6, 64), blk, 0, stream>>>(
      f_cat, tokin_w, tokin_b, f_h, 8192, 768, 1536, 768, nullptr, nullptr, 0, 0.f);

  // --- transformer blocks ---
  for (int i = 0; i < 8; ++i) {
    small_mm<true, false><<<72, blk, 0, stream>>>(
        f_temb, block_embed + (size_t)i * 768, cond1_w + (size_t)i * 768 * 2304,
        cond1_b + (size_t)i * 2304, f_c1, 8, 2304, 768);
    ln_mod_kernel<<<8192, blk, 0, stream>>>(
        f_h, ln1_w + (size_t)i * 768, ln1_b + (size_t)i * 768, f_c1, f_hn);
    gemm_bf16<EPI_PLAIN><<<dim3(18, 64), blk, 0, stream>>>(
        f_hn, qkv_w + (size_t)i * 768 * 2304, qkv_b + (size_t)i * 2304,
        f_qkv, 8192, 2304, 768, 2304, nullptr, nullptr, 0, 0.f);
    qkv_rope_kernel<<<24576, blk, 0, stream>>>(f_qkv, f_q, f_k, f_v);
    flash_attn<<<dim3(8, 96), blk, 0, stream>>>(f_q, f_k, f_v, f_y, 0.125f);
    gemm_bf16<EPI_GATERES><<<dim3(6, 64), blk, 0, stream>>>(
        f_y, attn_proj_w + (size_t)i * 768 * 768, attn_proj_b + (size_t)i * 768,
        f_h, 8192, 768, 768, 768, nullptr, f_c1 + 1536, 2304, RES_SCALE);
    small_mm<true, false><<<72, blk, 0, stream>>>(
        f_temb, block_embed + (size_t)i * 768, cond2_w + (size_t)i * 768 * 2304,
        cond2_b + (size_t)i * 2304, f_c2, 8, 2304, 768);
    ln_mod_kernel<<<8192, blk, 0, stream>>>(
        f_h, ln2_w + (size_t)i * 768, ln2_b + (size_t)i * 768, f_c2, f_hn);
    gemm_bf16<EPI_GELU><<<dim3(24, 64), blk, 0, stream>>>(
        f_hn, mlp_w1 + (size_t)i * 768 * 3072, mlp_b1 + (size_t)i * 3072,
        f_m1, 8192, 3072, 768, 3072, nullptr, nullptr, 0, 0.f);
    gemm_bf16<EPI_GATERES><<<dim3(6, 64), blk, 0, stream>>>(
        f_m1, mlp_w2 + (size_t)i * 3072 * 768, mlp_b2 + (size_t)i * 768,
        f_h, 8192, 768, 3072, 768, nullptr, f_c2 + 1536, 2304, RES_SCALE);
  }

  // --- unembed + un-patchify ---
  gemm_bf16<EPI_PLAIN><<<dim3(2, 64), blk, 0, stream>>>(
      f_h, unembed_w, unembed_b, f_pat, 8192, 256, 768, 256, nullptr, nullptr, 0, 0.f);
  final_out_kernel<<<8192, blk, 0, stream>>>(x, f_pat, out_gain, out);
}
